// PointNet2ASIS_46420006535339
// MI455X (gfx1250) — matrix operand order repaired and checked by reading, not compiled, not run on hardware
//
#include <hip/hip_runtime.h>

typedef __attribute__((ext_vector_type(16))) _Float16 v16h;
typedef __attribute__((ext_vector_type(8)))  float    v8f;

#define NPTS   4096
#define BATCH  2
#define NSAMP  32
#define KASIS  30
#define NUMCLS 13
#define INSCH  5

#define GEMM_WPB    4
#define FPS_THREADS 512
#define FPS_MAXN    4096

static inline int cdiv_h(int a, int b) { return (a + b - 1) / b; }
static inline int rup(int a, int b) { return ((a + b - 1) / b) * b; }

// ---------------------------------------------------------------------------
// Param index map: jax pytree flatten order (top-level keys sorted:
// coords, features, params; params keys sorted; each layer dict {'W','b'}
// flattens W then b). Leaves of params start at d_in[2].
// ---------------------------------------------------------------------------
enum ParamIdx : int {
  P_AD_W = 2, P_AD_B,
  P_IE_W, P_IE_B,
  P_INSFC_W, P_INSFC_B,
  P_INSFP1_W0, P_INSFP1_B0, P_INSFP1_W1, P_INSFP1_B1, P_INSFP1_W2, P_INSFP1_B2,
  P_INSFP2_W0, P_INSFP2_B0, P_INSFP2_W1, P_INSFP2_B1,
  P_INSFP3_W0, P_INSFP3_B0, P_INSFP3_W1, P_INSFP3_B1,
  P_INSFP4_W0, P_INSFP4_B0, P_INSFP4_W1, P_INSFP4_B1,
  P_SA1_W0, P_SA1_B0, P_SA1_W1, P_SA1_B1, P_SA1_W2, P_SA1_B2,
  P_SA2_W0, P_SA2_B0, P_SA2_W1, P_SA2_B1, P_SA2_W2, P_SA2_B2,
  P_SA3_W0, P_SA3_B0, P_SA3_W1, P_SA3_B1, P_SA3_W2, P_SA3_B2,
  P_SA4_W0, P_SA4_B0, P_SA4_W1, P_SA4_B1, P_SA4_W2, P_SA4_B2,
  P_SEMFC_W, P_SEMFC_B,
  P_SEMFP1_W0, P_SEMFP1_B0, P_SEMFP1_W1, P_SEMFP1_B1, P_SEMFP1_W2, P_SEMFP1_B2,
  P_SEMFP2_W0, P_SEMFP2_B0, P_SEMFP2_W1, P_SEMFP2_B1,
  P_SEMFP3_W0, P_SEMFP3_B0, P_SEMFP3_W1, P_SEMFP3_B1,
  P_SEMFP4_W0, P_SEMFP4_B0, P_SEMFP4_W1, P_SEMFP4_B1,
  P_SP_W, P_SP_B   // P_SP_B == 71, n_in == 72
};

struct WD { int idx, k, n; };
static const WD WTAB[] = {
  {P_AD_W,128,128},{P_IE_W,128,INSCH},{P_INSFC_W,128,128},
  {P_INSFP1_W0,134,128},{P_INSFP1_W1,128,128},{P_INSFP1_W2,128,128},
  {P_INSFP2_W0,320,256},{P_INSFP2_W1,256,128},
  {P_INSFP3_W0,384,256},{P_INSFP3_W1,256,256},
  {P_INSFP4_W0,768,256},{P_INSFP4_W1,256,256},
  {P_SA1_W0,9,32},{P_SA1_W1,32,32},{P_SA1_W2,32,64},
  {P_SA2_W0,67,64},{P_SA2_W1,64,64},{P_SA2_W2,64,128},
  {P_SA3_W0,131,128},{P_SA3_W1,128,128},{P_SA3_W2,128,256},
  {P_SA4_W0,259,256},{P_SA4_W1,256,256},{P_SA4_W2,256,512},
  {P_SEMFC_W,128,128},
  {P_SEMFP1_W0,134,128},{P_SEMFP1_W1,128,128},{P_SEMFP1_W2,128,128},
  {P_SEMFP2_W0,320,256},{P_SEMFP2_W1,256,128},
  {P_SEMFP3_W0,384,256},{P_SEMFP3_W1,256,256},
  {P_SEMFP4_W0,768,256},{P_SEMFP4_W1,256,256},
  {P_SP_W,128,NUMCLS},
};

// ===========================================================================
// Kernels
// ===========================================================================

// Pack f32 [K,N] weight into WMMA B-operand layout, f16, zero-padded:
// out[((tn*(Kp/32) + kb)*32 + lane)*16 + i] = W[kb*32 + (lane>>4)*16 + i, tn*16 + (lane&15)]
__global__ void k_pack_w(const float* __restrict__ W, int K, int N, int Kp,
                         _Float16* __restrict__ out) {
  int kblocks = Kp >> 5;
  int total = ((N + 15) >> 4) * kblocks * 512;
  int t = blockIdx.x * blockDim.x + threadIdx.x;
  if (t >= total) return;
  int i    = t & 15;
  int lane = (t >> 4) & 31;
  int rem  = t >> 9;
  int kb   = rem % kblocks;
  int tn   = rem / kblocks;
  int k    = kb * 32 + (lane >> 4) * 16 + i;
  int col  = tn * 16 + (lane & 15);
  float v = (k < K && col < N) ? W[(size_t)k * N + col] : 0.0f;
  out[t] = (_Float16)v;
}

// [C,N] channel-first -> [N,C] point-major
__global__ void k_transpose_cn(const float* __restrict__ s, int C, int N, float* __restrict__ d) {
  int t = blockIdx.x * blockDim.x + threadIdx.x;
  if (t >= C * N) return;
  int c = t / N, i = t - c * N;
  d[i * C + c] = s[t];
}

// [N,C] point-major -> [C,N] channel-first
__global__ void k_writeback_cf(const float* __restrict__ s, int C, int N, float* __restrict__ d) {
  int t = blockIdx.x * blockDim.x + threadIdx.x;
  if (t >= C * N) return;
  int i = t / C, c = t - i * C;
  d[c * N + i] = s[t];
}

__global__ void k_add(const float* __restrict__ a, const float* __restrict__ b,
                      float* __restrict__ c, int n) {
  int t = blockIdx.x * blockDim.x + threadIdx.x;
  if (t < n) c[t] = a[t] + b[t];
}

__global__ void k_gather_rows(const float* __restrict__ src, const int* __restrict__ idx,
                              int rows, int C, float* __restrict__ dst) {
  int t = blockIdx.x * blockDim.x + threadIdx.x;
  if (t >= rows * C) return;
  int r = t / C, c = t - r * C;
  dst[t] = src[idx[r] * C + c];
}

// dst[i*ld + c] = src[i*C1 + c]
__global__ void k_copy_cols(const float* __restrict__ src, int C1,
                            float* __restrict__ dst, int ld, int n) {
  int t = blockIdx.x * blockDim.x + threadIdx.x;
  if (t >= n * C1) return;
  int i = t / C1, c = t - i * C1;
  dst[(size_t)i * ld + c] = src[t];
}

// zero columns [c0, ld) of an [n, ld] row-major buffer
__global__ void k_zero_cols(float* __restrict__ buf, int ld, int c0, int n) {
  int width = ld - c0;
  int t = blockIdx.x * blockDim.x + threadIdx.x;
  if (t >= n * width) return;
  int i = t / width, c = c0 + (t - i * width);
  buf[(size_t)i * ld + c] = 0.0f;
}

// Farthest point sampling; exact sequential semantics of the reference scan.
__global__ __launch_bounds__(FPS_THREADS)
void k_fps(const float* __restrict__ xyz, int n, int npoint, int* __restrict__ out) {
  __shared__ float sd[FPS_MAXN];
  __shared__ float rv[FPS_THREADS];
  __shared__ int   ri[FPS_THREADS];
  __shared__ int   s_far;
  int tid = threadIdx.x;
  for (int i = tid; i < n; i += FPS_THREADS) sd[i] = 1e10f;
  if (tid == 0) s_far = 0;
  __syncthreads();
  for (int s = 0; s < npoint; ++s) {
    int far = s_far;
    if (tid == 0) out[s] = far;
    float fx = xyz[far * 3 + 0], fy = xyz[far * 3 + 1], fz = xyz[far * 3 + 2];
    float bv = -1.0f; int bi = 0;
    for (int i = tid; i < n; i += FPS_THREADS) {
      float dx = xyz[i * 3 + 0] - fx;
      float dy = xyz[i * 3 + 1] - fy;
      float dz = xyz[i * 3 + 2] - fz;
      float d  = dx * dx + dy * dy + dz * dz;
      float nd = fminf(sd[i], d);
      sd[i] = nd;
      if (nd > bv) { bv = nd; bi = i; }
    }
    rv[tid] = bv; ri[tid] = bi;
    __syncthreads();
    for (int off = FPS_THREADS / 2; off > 0; off >>= 1) {
      if (tid < off) {
        float ov = rv[tid + off]; int oi = ri[tid + off];
        if (ov > rv[tid] || (ov == rv[tid] && oi < ri[tid])) { rv[tid] = ov; ri[tid] = oi; }
      }
      __syncthreads();
    }
    if (tid == 0) s_far = ri[0];
    __syncthreads();
  }
}

// First nsample indices (ascending) with d2 <= r2; pad with first hit.
__global__ void k_ball(const float* __restrict__ xyz, int n,
                       const float* __restrict__ nxyz, int S,
                       float r2, int ns, int* __restrict__ out) {
  int s = blockIdx.x * blockDim.x + threadIdx.x;
  if (s >= S) return;
  float cx = nxyz[s * 3 + 0], cy = nxyz[s * 3 + 1], cz = nxyz[s * 3 + 2];
  int cnt = 0, first = 0;
  for (int j = 0; j < n && cnt < ns; ++j) {
    float dx = xyz[j * 3 + 0] - cx;
    float dy = xyz[j * 3 + 1] - cy;
    float dz = xyz[j * 3 + 2] - cz;
    float d2 = dx * dx + dy * dy + dz * dz;
    if (d2 <= r2) {
      if (cnt == 0) first = j;
      out[s * ns + cnt++] = j;
    }
  }
  for (; cnt < ns; ++cnt) out[s * ns + cnt] = first;
}

// g[s,j,:] = [xyz[g]-new_xyz[s] , feats[g], 0-pad]  (row-major [S*ns, ldp])
__global__ void k_group(const float* __restrict__ xyz, const float* __restrict__ feats, int Cf,
                        const float* __restrict__ nxyz, const int* __restrict__ gidx,
                        int S, int ns, float* __restrict__ out, int ldp) {
  int t = blockIdx.x * blockDim.x + threadIdx.x;
  if (t >= S * ns) return;
  int s = t / ns;
  int j = gidx[t];
  float* o = out + (size_t)t * ldp;
  o[0] = xyz[j * 3 + 0] - nxyz[s * 3 + 0];
  o[1] = xyz[j * 3 + 1] - nxyz[s * 3 + 1];
  o[2] = xyz[j * 3 + 2] - nxyz[s * 3 + 2];
  for (int c = 0; c < Cf; ++c) o[3 + c] = feats[j * Cf + c];
  for (int c = 3 + Cf; c < ldp; ++c) o[c] = 0.0f;
}

__global__ void k_maxpool(const float* __restrict__ in, int S, int ns, int C,
                          float* __restrict__ out) {
  int t = blockIdx.x * blockDim.x + threadIdx.x;
  if (t >= S * C) return;
  int s = t / C, c = t - s * C;
  float m = -3.4e38f;
  for (int j = 0; j < ns; ++j) m = fmaxf(m, in[((size_t)s * ns + j) * C + c]);
  out[t] = m;
}

// 3-NN inverse-distance interpolation; writes out[i*ld + coff + c]
__global__ void k_three_nn(const float* __restrict__ xyz1, int n,
                           const float* __restrict__ xyz2, int S,
                           const float* __restrict__ f2, int C2,
                           float* __restrict__ out, int ld, int coff) {
  int i = blockIdx.x * blockDim.x + threadIdx.x;
  if (i >= n) return;
  float px = xyz1[i * 3 + 0], py = xyz1[i * 3 + 1], pz = xyz1[i * 3 + 2];
  float b0 = 3.4e38f, b1 = 3.4e38f, b2 = 3.4e38f;
  int i0 = 0, i1 = 0, i2 = 0;
  for (int j = 0; j < S; ++j) {
    float dx = px - xyz2[j * 3 + 0];
    float dy = py - xyz2[j * 3 + 1];
    float dz = pz - xyz2[j * 3 + 2];
    float d = dx * dx + dy * dy + dz * dz;
    if (d < b0)      { b2 = b1; i2 = i1; b1 = b0; i1 = i0; b0 = d; i0 = j; }
    else if (d < b1) { b2 = b1; i2 = i1; b1 = d;  i1 = j; }
    else if (d < b2) { b2 = d;  i2 = j; }
  }
  float w0 = 1.0f / (fmaxf(b0, 0.0f) + 1e-8f);
  float w1 = 1.0f / (fmaxf(b1, 0.0f) + 1e-8f);
  float w2 = 1.0f / (fmaxf(b2, 0.0f) + 1e-8f);
  float ws = w0 + w1 + w2;
  w0 /= ws; w1 /= ws; w2 /= ws;
  const float* r0 = f2 + (size_t)i0 * C2;
  const float* r1 = f2 + (size_t)i1 * C2;
  const float* r2 = f2 + (size_t)i2 * C2;
  float* o = out + (size_t)i * ld + coff;
  for (int c = 0; c < C2; ++c) o[c] = r0[c] * w0 + r1[c] * w1 + r2[c] * w2;
}

// top-KASIS nearest neighbors in 5-dim embedding space (self included)
__global__ void k_knn(const float* __restrict__ e, int n, int* __restrict__ nn) {
  int i = blockIdx.x * blockDim.x + threadIdx.x;
  if (i >= n) return;
  float ei[INSCH];
#pragma unroll
  for (int c = 0; c < INSCH; ++c) ei[c] = e[i * INSCH + c];
  float bd[KASIS]; int bx[KASIS];
#pragma unroll
  for (int t = 0; t < KASIS; ++t) { bd[t] = 3.4e38f; bx[t] = 0; }
  for (int j = 0; j < n; ++j) {
    float d = 0.0f;
#pragma unroll
    for (int c = 0; c < INSCH; ++c) { float t = ei[c] - e[j * INSCH + c]; d += t * t; }
    if (d < bd[KASIS - 1]) {
      int t = KASIS - 1;
      while (t > 0 && bd[t - 1] > d) { bd[t] = bd[t - 1]; bx[t] = bx[t - 1]; --t; }
      bd[t] = d; bx[t] = j;
    }
  }
  for (int t = 0; t < KASIS; ++t) nn[i * KASIS + t] = bx[t];
}

__global__ void k_gather_max(const float* __restrict__ f, int C,
                             const int* __restrict__ nn, int n, int k,
                             float* __restrict__ out) {
  int t = blockIdx.x * blockDim.x + threadIdx.x;
  if (t >= n * C) return;
  int i = t / C, c = t - i * C;
  float m = -3.4e38f;
  for (int s = 0; s < k; ++s) m = fmaxf(m, f[(size_t)nn[i * k + s] * C + c]);
  out[t] = m;
}

// ---------------------------------------------------------------------------
// WMMA GEMM: C[M,N] = act(A[M,Kp(f32,padded)] * Wp(packed f16) + bias).
// Each wave computes a 16 x (NT*16) strip: one A fragment (4 aligned float4
// loads + packed cvt) is reused by NT consecutive n-tiles -> NT v_wmma per
// 32-K step. NT chosen so ntiles % NT == 0; packed B is zero-padded so no
// guards are needed inside the K loop.
// ---------------------------------------------------------------------------
template <int NT>
__global__ __launch_bounds__(32 * GEMM_WPB)
void k_gemm_wmma(const float* __restrict__ A, const _Float16* __restrict__ Wp,
                 const float* __restrict__ bias, float* __restrict__ C,
                 int M, int Kp, int N, int relu) {
  int ntiles  = (N + 15) >> 4;
  int ngroups = ntiles / NT;
  int groups  = (M >> 4) * ngroups;
  int g = blockIdx.x * GEMM_WPB + (threadIdx.x >> 5);
  if (g >= groups) return;
  int tm = g / ngroups;
  int tg = g - tm * ngroups;
  int lane = threadIdx.x & 31;
  int l16  = lane & 15;
  int half = lane >> 4;
  const float* arow = A + (size_t)(tm * 16 + l16) * Kp;
  size_t wstride = (size_t)Kp << 4;  // halfs per n-tile of packed B
  const _Float16* wp0 = Wp + (size_t)(tg * NT) * wstride + (lane << 4);
  v8f acc[NT] = {};
  for (int kb = 0; kb < Kp; kb += 32) {
    __builtin_prefetch(arow + kb + 32, 0, 0);
    __builtin_prefetch(wp0 + ((kb >> 5) + 1) * 512, 0, 0);
    const float4* pa = (const float4*)(arow + kb + half * 16);
    const float4* pb = (const float4*)(arow + kb + half * 16 + 8);
    float4 a0 = pa[0], a1 = pa[1];
    float4 a2 = pb[0], a3 = pb[1];
    v16h av;
    av[0]  = (_Float16)a0.x; av[1]  = (_Float16)a0.y;
    av[2]  = (_Float16)a0.z; av[3]  = (_Float16)a0.w;
    av[4]  = (_Float16)a1.x; av[5]  = (_Float16)a1.y;
    av[6]  = (_Float16)a1.z; av[7]  = (_Float16)a1.w;
    av[8]  = (_Float16)a2.x; av[9]  = (_Float16)a2.y;
    av[10] = (_Float16)a2.z; av[11] = (_Float16)a2.w;
    av[12] = (_Float16)a3.x; av[13] = (_Float16)a3.y;
    av[14] = (_Float16)a3.z; av[15] = (_Float16)a3.w;
    const _Float16* wk = wp0 + (size_t)(kb >> 5) * 512;
#pragma unroll
    for (int u = 0; u < NT; ++u) {
      v16h bv = *(const v16h*)(wk + (size_t)u * wstride);
      acc[u] = __builtin_amdgcn_wmma_f32_16x16x32_f16(
          /*neg_a=*/false, av, /*neg_b=*/false, bv,
          /*c_mod=*/(short)0, acc[u], /*reuse_a=*/false, /*reuse_b=*/false);
    }
  }
#pragma unroll
  for (int u = 0; u < NT; ++u) {
    int gn = (tg * NT + u) * 16 + l16;
    if (gn < N) {
      float bb = bias[gn];
      float* crow = C + (size_t)(tm * 16 + half * 8) * N + gn;
#pragma unroll
      for (int r = 0; r < 8; ++r) {
        float v = acc[u][r] + bb;
        if (relu) v = fmaxf(v, 0.0f);
        crow[(size_t)r * N] = v;
      }
    }
  }
}

// ===========================================================================
// Host orchestration
// ===========================================================================

static void launch_gemm(hipStream_t st, const float* A, const _Float16* Wp, const float* b,
                        float* C, int M, int Kp, int N, int relu) {
  int ntiles = cdiv_h(N, 16);
  int mt = cdiv_h(M, 16);
  if (ntiles % 4 == 0) {
    int groups = mt * (ntiles / 4);
    k_gemm_wmma<4><<<cdiv_h(groups, GEMM_WPB), 32 * GEMM_WPB, 0, st>>>(A, Wp, b, C, M, Kp, N, relu);
  } else if (ntiles % 2 == 0) {
    int groups = mt * (ntiles / 2);
    k_gemm_wmma<2><<<cdiv_h(groups, GEMM_WPB), 32 * GEMM_WPB, 0, st>>>(A, Wp, b, C, M, Kp, N, relu);
  } else {
    int groups = mt * ntiles;
    k_gemm_wmma<1><<<cdiv_h(groups, GEMM_WPB), 32 * GEMM_WPB, 0, st>>>(A, Wp, b, C, M, Kp, N, relu);
  }
}

extern "C" void kernel_launch(void* const* d_in, const int* in_sizes, int n_in,
                              void* d_out, int out_size, void* d_ws, size_t ws_size,
                              hipStream_t stream) {
  (void)in_sizes; (void)n_in; (void)out_size; (void)ws_size;
  const float* coords = (const float*)d_in[0];   // [B,3,N]
  const float* feats  = (const float*)d_in[1];   // [B,6,N]
  float* out = (float*)d_out;                    // [B,13,N] ++ [B,5,N]

  // ---- bump allocator over workspace ----
  char* wpb_ = (char*)d_ws;
  auto alloc = [&](size_t bytes) -> void* {
    void* r = (void*)wpb_;
    wpb_ += (bytes + 255) & ~(size_t)255;
    return r;
  };

  // ---- pack all weight matrices into WMMA B layout (f16, padded) ----
  _Float16* wpk[80] = {};
  int WN[80] = {}, WKP[80] = {};
  for (const WD& w : WTAB) {
    int Kp = rup(w.k, 32);
    int total = ((w.n + 15) / 16) * (Kp / 32) * 512;
    wpk[w.idx] = (_Float16*)alloc(sizeof(_Float16) * (size_t)total);
    WN[w.idx] = w.n; WKP[w.idx] = Kp;
    k_pack_w<<<cdiv_h(total, 256), 256, 0, stream>>>(
        (const float*)d_in[w.idx], w.k, w.n, Kp, wpk[w.idx]);
  }

  // ---- level geometry ----
  static const int   NL[5]  = {4096, 1024, 256, 64, 16};
  static const int   FCN[5] = {6, 64, 128, 256, 512};
  static const float R2[4]  = {0.01f, 0.04f, 0.16f, 0.64f};
  static const int   SA_W[4][3] = {
    {P_SA1_W0, P_SA1_W1, P_SA1_W2}, {P_SA2_W0, P_SA2_W1, P_SA2_W2},
    {P_SA3_W0, P_SA3_W1, P_SA3_W2}, {P_SA4_W0, P_SA4_W1, P_SA4_W2}};
  static const int FP_NLAY[4] = {2, 2, 2, 3};  // fp4, fp3, fp2, fp1
  static const int FP_W[2][4][3] = {
    {{P_SEMFP4_W0, P_SEMFP4_W1, -1}, {P_SEMFP3_W0, P_SEMFP3_W1, -1},
     {P_SEMFP2_W0, P_SEMFP2_W1, -1}, {P_SEMFP1_W0, P_SEMFP1_W1, P_SEMFP1_W2}},
    {{P_INSFP4_W0, P_INSFP4_W1, -1}, {P_INSFP3_W0, P_INSFP3_W1, -1},
     {P_INSFP2_W0, P_INSFP2_W1, -1}, {P_INSFP1_W0, P_INSFP1_W1, P_INSFP1_W2}}};

  // ---- buffers (reused across batches; stream is sequential) ----
  float* xs[5]; float* fs[5];
  for (int l = 0; l < 5; ++l) {
    xs[l] = (float*)alloc(sizeof(float) * (size_t)NL[l] * 3);
    fs[l] = (float*)alloc(sizeof(float) * (size_t)NL[l] * FCN[l]);
  }
  int*   fidx = (int*)alloc(sizeof(int) * 1024);
  int*   gidx = (int*)alloc(sizeof(int) * 1024 * NSAMP);
  float* ping = (float*)alloc(sizeof(float) * (size_t)32768 * 96);  // SA padded rows (ld<=96 at M=32768)
  float* pong = (float*)alloc(sizeof(float) * (size_t)32768 * 96);
  float* rb[3];
  for (int i = 0; i < 3; ++i) rb[i] = (float*)alloc(sizeof(float) * 700000);
  float* fbr[2];
  fbr[0] = (float*)alloc(sizeof(float) * (size_t)NPTS * 128);  // f_sem
  fbr[1] = (float*)alloc(sizeof(float) * (size_t)NPTS * 128);  // f_ins
  float* e_ins = (float*)alloc(sizeof(float) * (size_t)NPTS * INSCH);
  int*   nn    = (int*)alloc(sizeof(int) * (size_t)NPTS * KASIS);
  float* psem  = (float*)alloc(sizeof(float) * (size_t)NPTS * NUMCLS);

  for (int b = 0; b < BATCH; ++b) {
    // channel-first -> point-major
    k_transpose_cn<<<cdiv_h(3 * NPTS, 256), 256, 0, stream>>>(
        coords + (size_t)b * 3 * NPTS, 3, NPTS, xs[0]);
    k_transpose_cn<<<cdiv_h(6 * NPTS, 256), 256, 0, stream>>>(
        feats + (size_t)b * 6 * NPTS, 6, NPTS, fs[0]);

    // ---- Set abstraction levels ----
    for (int lv = 0; lv < 4; ++lv) {
      int n = NL[lv], S = NL[lv + 1], Cp = FCN[lv];
      k_fps<<<1, FPS_THREADS, 0, stream>>>(xs[lv], n, S, fidx);
      k_gather_rows<<<cdiv_h(S * 3, 256), 256, 0, stream>>>(xs[lv], fidx, S, 3, xs[lv + 1]);
      k_ball<<<cdiv_h(S, 128), 128, 0, stream>>>(xs[lv], n, xs[lv + 1], S, R2[lv], NSAMP, gidx);
      int ldp = rup(3 + Cp, 32);
      k_group<<<cdiv_h(S * NSAMP, 256), 256, 0, stream>>>(
          xs[lv], fs[lv], Cp, xs[lv + 1], gidx, S, NSAMP, ping, ldp);
      const float* a = ping; float* o = pong;
      int M = S * NSAMP, kp = ldp;
      int cout = 0;
      for (int l = 0; l < 3; ++l) {
        int wi = SA_W[lv][l];
        cout = WN[wi];
        launch_gemm(stream, a, wpk[wi], (const float*)d_in[wi + 1], o, M, kp, cout, 1);
        kp = cout;  // all SA couts are multiples of 32
        const float* t = a; a = o; o = (float*)t;
      }
      k_maxpool<<<cdiv_h(S * cout, 256), 256, 0, stream>>>(a, S, NSAMP, cout, fs[lv + 1]);
    }

    // ---- Feature propagation branches (0=sem, 1=ins) ----
    for (int pr = 0; pr < 2; ++pr) {
      const float* f2 = fs[4]; int C2 = FCN[4];
      int rot = 0;
      for (int j = 0; j < 4; ++j) {
        int lvl = 3 - j, n = NL[lvl], S = NL[lvl + 1], C1 = FCN[lvl];
        int cin = C1 + C2;
        int cinP = rup(cin, 32);
        float* cat = rb[rot]; rot = (rot + 1) % 3;
        k_copy_cols<<<cdiv_h(n * C1, 256), 256, 0, stream>>>(fs[lvl], C1, cat, cinP, n);
        k_three_nn<<<cdiv_h(n, 128), 128, 0, stream>>>(
            xs[lvl], n, xs[lvl + 1], S, f2, C2, cat, cinP, C1);
        if (cinP > cin)
          k_zero_cols<<<cdiv_h(n * (cinP - cin), 256), 256, 0, stream>>>(cat, cinP, cin, n);
        const float* a = cat; int kp = cinP;
        for (int l = 0; l < FP_NLAY[j]; ++l) {
          int wi = FP_W[pr][j][l];
          float* o = rb[rot]; rot = (rot + 1) % 3;
          launch_gemm(stream, a, wpk[wi], (const float*)d_in[wi + 1], o, n, kp, WN[wi], 1);
          a = o; kp = WN[wi];  // couts are multiples of 32
        }
        f2 = a; C2 = kp;
      }
      int fci = (pr == 0) ? P_SEMFC_W : P_INSFC_W;
      launch_gemm(stream, f2, wpk[fci], (const float*)d_in[fci + 1], fbr[pr], NPTS, 128, 128, 1);
    }

    // ---- ASIS head ----
    launch_gemm(stream, fbr[0], wpk[P_AD_W], (const float*)d_in[P_AD_B], rb[0],
                NPTS, 128, 128, 1);                                     // ad(f_sem)
    k_add<<<cdiv_h(NPTS * 128, 256), 256, 0, stream>>>(fbr[1], rb[0], rb[1], NPTS * 128);
    launch_gemm(stream, rb[1], wpk[P_IE_W], (const float*)d_in[P_IE_B], e_ins,
                NPTS, 128, INSCH, 0);                                   // e_ins
    k_knn<<<cdiv_h(NPTS, 128), 128, 0, stream>>>(e_ins, NPTS, nn);
    k_gather_max<<<cdiv_h(NPTS * 128, 256), 256, 0, stream>>>(fbr[0], 128, nn, NPTS, KASIS, rb[2]);
    launch_gemm(stream, rb[2], wpk[P_SP_W], (const float*)d_in[P_SP_B], psem,
                NPTS, 128, NUMCLS, 0);                                  // p_sem

    // ---- outputs, channel-first ----
    k_writeback_cf<<<cdiv_h(NPTS * NUMCLS, 256), 256, 0, stream>>>(
        psem, NUMCLS, NPTS, out + (size_t)b * NUMCLS * NPTS);
    k_writeback_cf<<<cdiv_h(NPTS * INSCH, 256), 256, 0, stream>>>(
        e_ins, INSCH, NPTS, out + (size_t)BATCH * NUMCLS * NPTS + (size_t)b * INSCH * NPTS);
  }
}
